// VLMamba_24859270709415
// MI455X (gfx1250) — matrix operand order repaired and checked by reading, not compiled, not run on hardware
//
#include <hip/hip_runtime.h>
#include <hip/hip_bf16.h>
#include <math.h>

// ---------------- model constants ----------------
#define D_MODEL   768
#define PATCH     16
#define IMG       224
#define IN_CH     3
#define VOCAB     50257
#define N_LAYERS  8
#define D_INNER   1536      // 2*768
#define D_STATE   16
#define D_CONV    4
#define DT_RANK   48
#define D_FF      3072      // 4*768
#define SEQ       512
#define NPATCH    196       // (224/16)^2
#define LSEQ      708       // NPATCH + SEQ
#define PATCH_DIM 768       // 3*16*16

typedef __attribute__((ext_vector_type(16))) _Float16 v16h;
typedef __attribute__((ext_vector_type(8)))  float    v8f;

#if __has_builtin(__builtin_amdgcn_global_load_async_to_lds_b64)
#define HAS_ASYNC_LDS 1
// param types per compiler diagnostic: vector_size(8) int, AS1 src / AS3 dst
typedef int v2i_t __attribute__((__vector_size__(8)));
typedef __attribute__((address_space(1))) v2i_t* gptr_v2i;
typedef __attribute__((address_space(3))) v2i_t* lptr_v2i;
#else
#define HAS_ASYNC_LDS 0
#endif

__device__ __forceinline__ void wait_async_zero()
{
#if HAS_ASYNC_LDS
#if __has_builtin(__builtin_amdgcn_s_wait_asynccnt)
    __builtin_amdgcn_s_wait_asynccnt(0);
#else
    asm volatile("s_wait_asynccnt 0" ::: "memory");
#endif
#endif
}

// ---------------- small device helpers ----------------
__device__ __forceinline__ float silu_f(float x)     { return x / (1.0f + expf(-x)); }
__device__ __forceinline__ float gelu_f(float x)     { return 0.5f * x * (1.0f + erff(x * 0.70710678118654752f)); }
__device__ __forceinline__ float softplus_f(float x) { return (x > 20.0f) ? x : log1pf(expf(x)); }

// =====================================================================
// Generic WMMA GEMM:  C[M,N] = act( A[M,K] @ W[N,K]^T + bias + resid )
// Block = 8 waves; one 16x32 f32 A tile staged in LDS (async DMA when
// available); each wave computes TWO 16x16 tiles (1 A frag, 2 B frags).
// act: 0 = none, 1 = exact GELU, 2 = softplus
// =====================================================================
#define AF_STRIDE 36   // floats per LDS row: 144B rows keep f4/b64 stores aligned

// B fragment: W row-major [N,K]; B(k,n) = W(n,k).
__device__ __forceinline__ v16h load_b_frag(const float* __restrict__ W, int ldw,
                                            int N, int K, int n0, int k0, int lane)
{
    const int n  = n0 + (lane & 15);
    const int kc = k0 + ((lane >> 4) << 4);
    const float* wp = W + (size_t)n * ldw + kc;
    v16h b;
    if ((n < N) && (kc + 16 <= K)) {
        __builtin_prefetch(wp + 32, 0, 1);     // next K tile of this weight row
#pragma unroll
        for (int i = 0; i < 16; i += 4) {
            const float4 f = *reinterpret_cast<const float4*>(wp + i);
            b[i + 0] = (_Float16)f.x;
            b[i + 1] = (_Float16)f.y;
            b[i + 2] = (_Float16)f.z;
            b[i + 3] = (_Float16)f.w;
        }
    } else {
#pragma unroll
        for (int i = 0; i < 16; ++i) {
            float v = (n < N && (kc + i) < K) ? wp[i] : 0.0f;
            b[i] = (_Float16)v;
        }
    }
    return b;
}

__device__ __forceinline__ void store_tile(float* __restrict__ C, int ldc,
                                           int M, int N, int m0, int n0,
                                           const v8f acc,
                                           const float* __restrict__ bias,
                                           const float* __restrict__ resid, int ldr,
                                           int act, int lane)
{
    const int col = n0 + (lane & 15);
    if (col >= N) return;
    const int rOff = (lane >> 4) << 3;        // 0 or 8
#pragma unroll
    for (int r = 0; r < 8; ++r) {
        const int row = m0 + r + rOff;
        if (row < M) {
            float v = acc[r];
            if (bias)  v += bias[col];
            if (resid) v += resid[(size_t)row * ldr + col];
            if (act == 1)      v = gelu_f(v);
            else if (act == 2) v = softplus_f(v);
            C[(size_t)row * ldc + col] = v;
        }
    }
}

__global__ __launch_bounds__(256)
void gemm_wmma_f16(const float* __restrict__ A, int lda,
                   const float* __restrict__ W, int ldw,
                   const float* __restrict__ bias,
                   const float* __restrict__ resid, int ldr,
                   float* __restrict__ C, int ldc,
                   int M, int N, int K, int act)
{
    __shared__ float aTileF[16 * AF_STRIDE];   // f32 staging (raw copy target)

    const int lane = threadIdx.x & 31;
    const int wid  = threadIdx.x >> 5;

    const int m0 = blockIdx.y << 4;
    const int ntiles = (N + 15) >> 4;
    int ta = (blockIdx.x * 8 + wid) * 2;
    int tb = ta + 1;
    if (ta > ntiles - 1) ta = ntiles - 1;     // clamped duplicates store identical data
    if (tb > ntiles - 1) tb = ntiles - 1;
    const int n0a = ta << 4;
    const int n0b = tb << 4;

    // A fragment coords (contiguous k = 16*(lane/16) + i, same order as B; 16-bit A 16x32, wave32)
    const int aRow = lane & 15;
    const int aKb  = (lane >> 4) << 4;        // 0 or 16

    const bool mFull = (m0 + 16 <= M);

    v8f acc0 = {};
    v8f acc1 = {};

    for (int k0 = 0; k0 < K; k0 += 32) {
        const bool kFull = (k0 + 32 <= K);

        // ---- stage A tile (16x32 f32) into LDS ----
        if (mFull && kFull) {
#if HAS_ASYNC_LDS
            // 256 threads x async b64 (2 floats) = full 2KB tile, ASYNCcnt-tracked DMA
            {
                const int r = threadIdx.x >> 4;
                const int c = (threadIdx.x & 15) << 1;
                const float* gp = A + (size_t)(m0 + r) * lda + k0 + c;
                float*       lp = &aTileF[r * AF_STRIDE + c];
                __builtin_amdgcn_global_load_async_to_lds_b64(
                    (gptr_v2i)(uintptr_t)gp,
                    (lptr_v2i)(uint32_t)(uintptr_t)lp,
                    0, 0);
            }
            wait_async_zero();
#else
            if (threadIdx.x < 128) {          // 128 threads x float4 = 512 floats
                const int r = threadIdx.x >> 3;
                const int c = (threadIdx.x & 7) << 2;
                const float4 f = *reinterpret_cast<const float4*>(
                    A + (size_t)(m0 + r) * lda + k0 + c);
                *reinterpret_cast<float4*>(&aTileF[r * AF_STRIDE + c]) = f;
            }
#endif
        } else {
            for (int idx = threadIdx.x; idx < 16 * 32; idx += 256) {
                const int r = idx >> 5, c = idx & 31;
                const int row = m0 + r, col = k0 + c;
                aTileF[r * AF_STRIDE + c] =
                    (row < M && col < K) ? A[(size_t)row * lda + col] : 0.0f;
            }
        }
        __syncthreads();

        // ---- A fragment from LDS (two 8-float groups per lane), cvt to f16 ----
        v16h af;
        {
            const float* ap = &aTileF[aRow * AF_STRIDE + aKb];
#pragma unroll
            for (int i = 0; i < 8; ++i) af[i]     = (_Float16)ap[i];
#pragma unroll
            for (int i = 0; i < 8; ++i) af[8 + i] = (_Float16)ap[8 + i];
        }

        // ---- two B fragments, two WMMAs ----
        const v16h bfa = load_b_frag(W, ldw, N, K, n0a, k0, lane);
        const v16h bfb = load_b_frag(W, ldw, N, K, n0b, k0, lane);

        acc0 = __builtin_amdgcn_wmma_f32_16x16x32_f16(
            false, af, false, bfa, (short)0, acc0, false, false);
        acc1 = __builtin_amdgcn_wmma_f32_16x16x32_f16(
            false, af, false, bfb, (short)0, acc1, false, false);

        __syncthreads();
    }

    store_tile(C, ldc, M, N, m0, n0a, acc0, bias, resid, ldr, act, lane);
    if (tb != ta)
        store_tile(C, ldc, M, N, m0, n0b, acc1, bias, resid, ldr, act, lane);
}

// =====================================================================
// Patch gather: image [3,224,224] -> patches [196, 768]
// feature index = ph*48 + pw*3 + c  (b, py, px, ph, pw, c ordering)
// =====================================================================
__global__ __launch_bounds__(256)
void gather_patches(const float* __restrict__ image, float* __restrict__ patches)
{
    int idx = blockIdx.x * 256 + threadIdx.x;
    if (idx >= NPATCH * PATCH_DIM) return;
    int p = idx / PATCH_DIM;
    int f = idx - p * PATCH_DIM;
    int py = p / 14, px = p - py * 14;
    int ph = f / 48;
    int rem = f - ph * 48;
    int pw = rem / 3;
    int c  = rem - pw * 3;
    int y = py * PATCH + ph;
    int x = px * PATCH + pw;
    patches[idx] = image[((size_t)c * IMG + y) * IMG + x];
}

// =====================================================================
// Token embedding: x[196+t, :] = embed[ids[t], :] * sqrt(768)
// =====================================================================
__global__ __launch_bounds__(256)
void embed_tokens(const float* __restrict__ embed, const int* __restrict__ ids,
                  float* __restrict__ x)
{
    int t = blockIdx.x;
    int id = ids[t];
    const float scale = 27.712812921102035f;   // sqrt(768)
    const float* src = embed + (size_t)id * D_MODEL;
    float* dst = x + (size_t)(NPATCH + t) * D_MODEL;
    for (int j = threadIdx.x; j < D_MODEL; j += 256)
        dst[j] = src[j] * scale;
}

// =====================================================================
// LayerNorm over last dim D (one block per row, in-place safe)
// =====================================================================
__global__ __launch_bounds__(256)
void layernorm_kernel(const float* __restrict__ in, float* __restrict__ out,
                      const float* __restrict__ g, const float* __restrict__ b, int D)
{
    __shared__ float red[256];
    const int row = blockIdx.x;
    const float* p = in + (size_t)row * D;

    float s = 0.0f;
    for (int j = threadIdx.x; j < D; j += 256) s += p[j];
    red[threadIdx.x] = s; __syncthreads();
    for (int off = 128; off > 0; off >>= 1) {
        if (threadIdx.x < off) red[threadIdx.x] += red[threadIdx.x + off];
        __syncthreads();
    }
    const float mu = red[0] / (float)D; __syncthreads();

    float v = 0.0f;
    for (int j = threadIdx.x; j < D; j += 256) { float d = p[j] - mu; v += d * d; }
    red[threadIdx.x] = v; __syncthreads();
    for (int off = 128; off > 0; off >>= 1) {
        if (threadIdx.x < off) red[threadIdx.x] += red[threadIdx.x + off];
        __syncthreads();
    }
    const float inv = rsqrtf(red[0] / (float)D + 1e-5f); __syncthreads();

    for (int j = threadIdx.x; j < D; j += 256)
        out[(size_t)row * D + j] = (p[j] - mu) * inv * g[j] + b[j];
}

// =====================================================================
// Causal depthwise conv1d (k=4) + SiLU:  xi[l,d] from xz[:, 0:1536]
// =====================================================================
__global__ __launch_bounds__(256)
void conv_silu(const float* __restrict__ xz, const float* __restrict__ cw,
               const float* __restrict__ cb, float* __restrict__ xi)
{
    int idx = blockIdx.x * 256 + threadIdx.x;
    if (idx >= LSEQ * D_INNER) return;
    int l = idx / D_INNER;
    int d = idx - l * D_INNER;
    float acc = cb[d];
#pragma unroll
    for (int j = 0; j < D_CONV; ++j) {
        int ll = l - (D_CONV - 1) + j;
        if (ll >= 0) acc += cw[d * D_CONV + j] * xz[(size_t)ll * (2 * D_INNER) + d];
    }
    xi[idx] = silu_f(acc);
}

// =====================================================================
// Selective scan: one thread per channel d, 16-state recurrence in regs
// y[l,d] = (sum_s h[l,d,s]*C[l,s] + D[d]*xi[l,d]) * silu(z[l,d])
// =====================================================================
__global__ __launch_bounds__(256)
void mamba_scan(const float* __restrict__ xi, const float* __restrict__ xz,
                const float* __restrict__ dt, const float* __restrict__ xdb,
                const float* __restrict__ a_log, const float* __restrict__ dpar,
                float* __restrict__ y)
{
    int d = blockIdx.x * 256 + threadIdx.x;
    if (d >= D_INNER) return;

    float A[D_STATE], h[D_STATE];
#pragma unroll
    for (int s = 0; s < D_STATE; ++s) {
        A[s] = -expf(a_log[d * D_STATE + s]);
        h[s] = 0.0f;
    }
    const float Dv = dpar[d];

    for (int l = 0; l < LSEQ; ++l) {
        const float dtv = dt[(size_t)l * D_INNER + d];
        const float xiv = xi[(size_t)l * D_INNER + d];
        const float zv  = xz[(size_t)l * (2 * D_INNER) + D_INNER + d];
        const float* bc = xdb + (size_t)l * (DT_RANK + 2 * D_STATE);
        float accy = 0.0f;
#pragma unroll
        for (int s = 0; s < D_STATE; ++s) {
            float dA = expf(dtv * A[s]);
            float Bv = bc[DT_RANK + s];
            float Cv = bc[DT_RANK + D_STATE + s];
            h[s] = dA * h[s] + dtv * Bv * xiv;
            accy += h[s] * Cv;
        }
        y[(size_t)l * D_INNER + d] = (accy + Dv * xiv) * silu_f(zv);
    }
}

// =====================================================================
// Host-side orchestration
// =====================================================================
static inline void launch_gemm(const float* A, int lda, const float* W, int ldw,
                               const float* bias, const float* resid, int ldr,
                               float* C, int ldc, int M, int N, int K, int act,
                               hipStream_t s)
{
    dim3 grid((N + 255) / 256, (M + 15) / 16);   // 8 waves x 2 tiles = 256 cols/block
    gemm_wmma_f16<<<grid, 256, 0, s>>>(A, lda, W, ldw, bias, resid, ldr, C, ldc, M, N, K, act);
}

extern "C" void kernel_launch(void* const* d_in, const int* in_sizes, int n_in,
                              void* d_out, int out_size, void* d_ws, size_t ws_size,
                              hipStream_t stream)
{
    (void)in_sizes; (void)n_in; (void)out_size; (void)ws_size;

    const float* image    = (const float*)d_in[0];
    const int*   ids      = (const int*)  d_in[1];
    const float* proj_w   = (const float*)d_in[2];
    const float* proj_b   = (const float*)d_in[3];
    const float* pos_emb  = (const float*)d_in[4];
    const float* br_w1    = (const float*)d_in[5];
    const float* br_b1    = (const float*)d_in[6];
    const float* br_w2    = (const float*)d_in[7];
    const float* br_b2    = (const float*)d_in[8];
    const float* br_ln_g  = (const float*)d_in[9];
    const float* br_ln_b  = (const float*)d_in[10];
    const float* embed    = (const float*)d_in[11];
    const float* in_projw = (const float*)d_in[12];
    const float* conv_w   = (const float*)d_in[13];
    const float* conv_b   = (const float*)d_in[14];
    const float* x_projw  = (const float*)d_in[15];
    const float* dt_w     = (const float*)d_in[16];
    const float* dt_b     = (const float*)d_in[17];
    const float* a_log    = (const float*)d_in[18];
    const float* d_par    = (const float*)d_in[19];
    const float* norm_g   = (const float*)d_in[20];
    const float* norm_b   = (const float*)d_in[21];
    const float* out_w    = (const float*)d_in[22];

    float* out = (float*)d_out;
    float* ws  = (float*)d_ws;

    // workspace layout (floats)
    size_t o = 0;
    auto alloc = [&](size_t n) { float* p = ws + o; o += (n + 63) & ~(size_t)63; return p; };
    float* x       = alloc((size_t)LSEQ * D_MODEL);
    float* x_ln    = alloc((size_t)LSEQ * D_MODEL);
    float* patches = alloc((size_t)NPATCH * PATCH_DIM);
    float* h_ff    = alloc((size_t)NPATCH * D_FF);
    float* xz      = alloc((size_t)LSEQ * 2 * D_INNER);
    float* xi      = alloc((size_t)LSEQ * D_INNER);
    float* dtb     = alloc((size_t)LSEQ * D_INNER);
    float* xdb     = alloc((size_t)LSEQ * (DT_RANK + 2 * D_STATE));
    float* yb      = alloc((size_t)LSEQ * D_INNER);

    // ---- vision branch ----
    gather_patches<<<(NPATCH * PATCH_DIM + 255) / 256, 256, 0, stream>>>(image, patches);

    // vis = patches @ proj_w^T + proj_b + pos_embed     -> x[0:196]
    launch_gemm(patches, PATCH_DIM, proj_w, PATCH_DIM, proj_b, pos_emb, D_MODEL,
                x, D_MODEL, NPATCH, D_MODEL, PATCH_DIM, /*act*/0, stream);
    // h = gelu(vis @ br_w1^T + br_b1)
    launch_gemm(x, D_MODEL, br_w1, D_MODEL, br_b1, nullptr, 0,
                h_ff, D_FF, NPATCH, D_FF, D_MODEL, /*act*/1, stream);
    // vis = LN(h @ br_w2^T + br_b2)
    launch_gemm(h_ff, D_FF, br_w2, D_FF, br_b2, nullptr, 0,
                x, D_MODEL, NPATCH, D_MODEL, D_FF, /*act*/0, stream);
    layernorm_kernel<<<NPATCH, 256, 0, stream>>>(x, x, br_ln_g, br_ln_b, D_MODEL);

    // ---- text branch: x[196:708] ----
    embed_tokens<<<SEQ, 256, 0, stream>>>(embed, ids, x);

    // ---- Mamba layers ----
    for (int i = 0; i < N_LAYERS; ++i) {
        const float* in_w  = in_projw + (size_t)i * 2 * D_INNER * D_MODEL;
        const float* cw    = conv_w   + (size_t)i * D_INNER * D_CONV;
        const float* cb    = conv_b   + (size_t)i * D_INNER;
        const float* xw    = x_projw  + (size_t)i * (DT_RANK + 2 * D_STATE) * D_INNER;
        const float* dtw   = dt_w     + (size_t)i * D_INNER * DT_RANK;
        const float* dtbi  = dt_b     + (size_t)i * D_INNER;
        const float* al    = a_log    + (size_t)i * D_INNER * D_STATE;
        const float* dpi   = d_par    + (size_t)i * D_INNER;
        const float* ow    = out_w    + (size_t)i * D_MODEL * D_INNER;

        // xz = x @ in_w^T
        launch_gemm(x, D_MODEL, in_w, D_MODEL, nullptr, nullptr, 0,
                    xz, 2 * D_INNER, LSEQ, 2 * D_INNER, D_MODEL, 0, stream);
        // xi = silu(causal_conv(xz[:, :1536]))
        conv_silu<<<(LSEQ * D_INNER + 255) / 256, 256, 0, stream>>>(xz, cw, cb, xi);
        // xdb = xi @ xw^T       [708, 80]
        launch_gemm(xi, D_INNER, xw, D_INNER, nullptr, nullptr, 0,
                    xdb, DT_RANK + 2 * D_STATE, LSEQ, DT_RANK + 2 * D_STATE, D_INNER, 0, stream);
        // dt = softplus(xdb[:, :48] @ dtw^T + dtb)      [708, 1536]
        launch_gemm(xdb, DT_RANK + 2 * D_STATE, dtw, DT_RANK, dtbi, nullptr, 0,
                    dtb, D_INNER, LSEQ, D_INNER, DT_RANK, /*act*/2, stream);
        // selective scan
        mamba_scan<<<(D_INNER + 255) / 256, 256, 0, stream>>>(xi, xz, dtb, xdb, al, dpi, yb);
        // x = y @ ow^T
        launch_gemm(yb, D_INNER, ow, D_INNER, nullptr, nullptr, 0,
                    x, D_MODEL, LSEQ, D_MODEL, D_INNER, 0, stream);
    }

    // ---- final LN + vocab projection ----
    layernorm_kernel<<<LSEQ, 256, 0, stream>>>(x, x_ln, norm_g, norm_b, D_MODEL);
    launch_gemm(x_ln, D_MODEL, embed, D_MODEL, nullptr, nullptr, 0,
                out, VOCAB, LSEQ, VOCAB, D_MODEL, 0, stream);
}
